// Graph_Converter_17540646437051
// MI455X (gfx1250) — compile-verified, matrix operand order checked by reading, not run on hardware
//
#include <hip/hip_runtime.h>
#include <hip/hip_bf16.h>
#include <math.h>

// ---------------------------------------------------------------------------
// MI455X / gfx1250: wave32, WMMA 16x16x32 bf16 (f32 accum)
// ---------------------------------------------------------------------------
typedef __bf16 bf16_t;
typedef __attribute__((ext_vector_type(16))) __bf16 v16bf;
typedef __attribute__((ext_vector_type(8)))  float  v8f;

#define NSEQ  4096
#define DIMM  256
#define NHEAD 8
#define DHD   32
#define KTOP  32

__device__ __forceinline__ bf16_t f2bf(float f) {
  union { float f; unsigned u; } x; x.f = f;
  unsigned r = (x.u + 0x7FFFu + ((x.u >> 16) & 1u)) >> 16;
  union { unsigned short s; bf16_t b; } y; y.s = (unsigned short)r;
  return y.b;
}

// Load one 16x32 bf16 WMMA operand fragment from a row-major matrix.
// Lanes 0-15: row (row0+lane), K elems {0..7,16..23};
// lanes 16-31: row (row0+lane-16), K elems {8..15,24..31}.
// Two contiguous 16-byte loads per lane -> global_load_b128 pairs.
__device__ __forceinline__ v16bf load_frag16x32(const bf16_t* __restrict__ base,
                                                int ld, int row0, int k0, int lane) {
  int r  = row0 + (lane & 15);
  int hi = (lane >> 4) & 1;
  const bf16_t* p = base + (size_t)r * ld + k0 + hi * 8;
  union { uint4 u[2]; v16bf v; } u;
  u.u[0] = *(const uint4*)(p);
  u.u[1] = *(const uint4*)(p + 16);
  return u.v;
}

__device__ __forceinline__ float gelu_tanh(float v) {
  return 0.5f * v * (1.0f + tanhf(0.7978845608028654f * (v + 0.044715f * v * v * v)));
}

// ---------------------------------------------------------------------------
// Weight prep: f32 [K][N] -> bf16 transposed [N][K]
// ---------------------------------------------------------------------------
__global__ __launch_bounds__(256) void transpose_bf16_kernel(
    const float* __restrict__ W, bf16_t* __restrict__ Wt, int K, int Nc) {
  int i = blockIdx.x * blockDim.x + threadIdx.x;
  if (i >= K * Nc) return;
  int k = i / Nc, n = i % Nc;
  Wt[(size_t)n * K + k] = f2bf(W[(size_t)k * Nc + n]);
}

// ---------------------------------------------------------------------------
// Sigmoid-scaled RMSNorm: xn = rmsnorm(sigmoid(gamma)*x) * w  -> bf16
// ---------------------------------------------------------------------------
__global__ __launch_bounds__(256) void rmsnorm_kernel(
    const float* __restrict__ x, const float* __restrict__ w,
    const float* __restrict__ gamma_p, bf16_t* __restrict__ xn) {
  __shared__ float sred[256];
  int row = blockIdx.x, tid = threadIdx.x;
  float g  = 1.0f / (1.0f + __expf(-gamma_p[0]));
  float xv = g * x[(size_t)row * DIMM + tid];
  sred[tid] = xv * xv;
  __syncthreads();
  for (int s = 128; s > 0; s >>= 1) {
    if (tid < s) sred[tid] += sred[tid + s];
    __syncthreads();
  }
  float rms = rsqrtf(sred[0] * (1.0f / DIMM) + 1e-6f);
  xn[(size_t)row * DIMM + tid] = f2bf(xv * rms * w[tid]);
}

// ---------------------------------------------------------------------------
// WMMA GEMM, register-blocked 32x64 per wave (2 M-tiles x 4 N-tiles):
//   per k-step: 2 A-frags + 4 B-frags feed 8 WMMAs (0.75KB loaded per WMMA).
// C[M][Nt] = A[M][Kd] x Bt[Nt][Kd]^T, f32 accum; optional bias/gelu epilogue.
// ---------------------------------------------------------------------------
__global__ __launch_bounds__(256) void gemm_bf16_kernel(
    const bf16_t* __restrict__ A, const bf16_t* __restrict__ Bt,
    const float* __restrict__ bias, float* __restrict__ Cf,
    bf16_t* __restrict__ Cb, int M, int Kd, int Nt, int act) {
  int lane = threadIdx.x & 31;
  int wave = (blockIdx.x * blockDim.x + threadIdx.x) >> 5;
  int ntn  = Nt >> 6;                       // 64-wide N blocks
  if (wave >= (M >> 5) * ntn) return;       // wave-uniform exit
  int m0 = (wave / ntn) << 5;
  int n0 = (wave % ntn) << 6;

  v8f c[2][4] = {};
  for (int k0 = 0; k0 < Kd; k0 += 32) {
    v16bf a0 = load_frag16x32(A, Kd, m0,      k0, lane);
    v16bf a1 = load_frag16x32(A, Kd, m0 + 16, k0, lane);
#pragma unroll
    for (int j = 0; j < 4; ++j) {
      v16bf b = load_frag16x32(Bt, Kd, n0 + (j << 4), k0, lane);
      c[0][j] = __builtin_amdgcn_wmma_f32_16x16x32_bf16(false, a0, false, b,
                                                        (short)0, c[0][j], false, false);
      c[1][j] = __builtin_amdgcn_wmma_f32_16x16x32_bf16(false, a1, false, b,
                                                        (short)0, c[1][j], false, false);
    }
  }
  // C layout per tile: VGPR r -> row r (lanes 0-15) / 8+r (lanes 16-31), col lane&15
  int rsub = (lane >> 4) << 3;
#pragma unroll
  for (int j = 0; j < 4; ++j) {
    int col = n0 + (j << 4) + (lane & 15);
    float bcol = bias ? bias[col] : 0.0f;
#pragma unroll
    for (int i = 0; i < 2; ++i) {
      int rbase = m0 + (i << 4) + rsub;
#pragma unroll
      for (int r = 0; r < 8; ++r) {
        float v = c[i][j][r] + bcol;
        if (act) v = gelu_tanh(v);
        size_t o = (size_t)(rbase + r) * Nt + col;
        if (Cf) Cf[o] = v;
        if (Cb) Cb[o] = f2bf(v);
      }
    }
  }
}

// ---------------------------------------------------------------------------
// Fused scores + top-k + softmax + V-gather.
// One wave per (head, 32 query rows): each K-fragment feeds 2 WMMAs, and the
// top-k scan maps row<->lane with all 32 wave32 lanes active.
// Tasks are head-innermost so the 64MB bias matrix is served from L2 for 7/8 heads.
// ---------------------------------------------------------------------------
__global__ __launch_bounds__(128) void attn_topk_kernel(
    const bf16_t* __restrict__ Qb, const bf16_t* __restrict__ Kb,
    const float* __restrict__ Vf, const float* __restrict__ bias,
    bf16_t* __restrict__ attn_out, int* __restrict__ idx_out) {
  constexpr int WPB = 4;
  __shared__ float s_tile[WPB][32][17];
  __shared__ float s_val [WPB][32][KTOP];
  __shared__ int   s_idx [WPB][32][KTOP];
  __shared__ float s_w   [WPB][32][KTOP];

  int lane = threadIdx.x & 31;
  int w    = threadIdx.x >> 5;
  int task = blockIdx.x * WPB + w;
  int h    = task & (NHEAD - 1);
  int m0   = (task >> 3) << 5;              // 32 query rows per wave
  const float scale = 0.17677669529663687f; // 1/sqrt(32)

#pragma unroll
  for (int k = 0; k < KTOP; ++k) { s_val[w][lane][k] = -INFINITY; s_idx[w][lane][k] = 0; }

  const bf16_t* Qh = Qb + h * DHD;
  const bf16_t* Kh = Kb + h * DHD;
  v16bf aq0 = load_frag16x32(Qh, DIMM, m0,      0, lane);
  v16bf aq1 = load_frag16x32(Qh, DIMM, m0 + 16, 0, lane);

  float curmin = -INFINITY; int argmin = 0;  // per lane == per query row
  for (int ct = 0; ct < NSEQ / 16; ++ct) {
    v16bf bk = load_frag16x32(Kh, DIMM, ct << 4, 0, lane);
    if (ct + 1 < NSEQ / 16)  // prefetch next K tile (global_prefetch_b8)
      __builtin_prefetch(Kh + (size_t)((ct + 1) << 4) * DIMM + (lane << 3), 0, 1);
    v8f c0 = {}, c1 = {};
    c0 = __builtin_amdgcn_wmma_f32_16x16x32_bf16(false, aq0, false, bk, (short)0, c0,
                                                 false, false);
    c1 = __builtin_amdgcn_wmma_f32_16x16x32_bf16(false, aq1, false, bk, (short)0, c1,
                                                 false, false);
    int colg = (ct << 4) + (lane & 15);
    int rb   = (lane >> 4) << 3;
#pragma unroll
    for (int r = 0; r < 8; ++r) {
      s_tile[w][rb + r][lane & 15] =
          c0[r] * scale + bias[(size_t)(m0 + rb + r) * NSEQ + colg];
      s_tile[w][16 + rb + r][lane & 15] =
          c1[r] * scale + bias[(size_t)(m0 + 16 + rb + r) * NSEQ + colg];
    }
    __syncthreads();
    // top-k maintenance: lane = query row (all 32 lanes busy)
    for (int j = 0; j < 16; ++j) {
      float vv = s_tile[w][lane][j];
      if (vv > curmin) {
        s_val[w][lane][argmin] = vv;
        s_idx[w][lane][argmin] = (ct << 4) + j;
        curmin = s_val[w][lane][0]; argmin = 0;
#pragma unroll
        for (int k = 1; k < KTOP; ++k) {
          float t = s_val[w][lane][k];
          if (t < curmin) { curmin = t; argmin = k; }
        }
      }
    }
    __syncthreads();
  }

  // sort descending (top_k returns sorted) + softmax over the 32 selected
  {
    for (int i = 1; i < KTOP; ++i) {
      float v0 = s_val[w][lane][i]; int i0 = s_idx[w][lane][i];
      int j = i - 1;
      while (j >= 0 && s_val[w][lane][j] < v0) {
        s_val[w][lane][j + 1] = s_val[w][lane][j];
        s_idx[w][lane][j + 1] = s_idx[w][lane][j];
        --j;
      }
      s_val[w][lane][j + 1] = v0; s_idx[w][lane][j + 1] = i0;
    }
    float mx = s_val[w][lane][0], sum = 0.0f;
#pragma unroll
    for (int k = 0; k < KTOP; ++k) {
      float e = __expf(s_val[w][lane][k] - mx);
      s_w[w][lane][k] = e; sum += e;
    }
    float inv = 1.0f / sum;
#pragma unroll
    for (int k = 0; k < KTOP; ++k) s_w[w][lane][k] *= inv;
  }
  __syncthreads();

  // emit topk indices ([H][N][K] int32): lane k writes entry k of each row
  for (int r = 0; r < 32; ++r)
    idx_out[((size_t)h * NSEQ + (m0 + r)) * KTOP + lane] = s_idx[w][r][lane];

  // weighted V gather: lane = head-dim channel (coalesced 128B per neighbor row)
  for (int r = 0; r < 32; ++r) {
    float acc = 0.0f;
#pragma unroll
    for (int k = 0; k < KTOP; ++k)
      acc += s_w[w][r][k] * Vf[(size_t)s_idx[w][r][k] * DIMM + h * DHD + lane];
    attn_out[(size_t)(m0 + r) * DIMM + h * DHD + lane] = f2bf(acc);
  }
}

// ---------------------------------------------------------------------------
// out = sigmoid(alpha)*ffn + sigmoid(beta)*attn
// ---------------------------------------------------------------------------
__global__ __launch_bounds__(256) void combine_kernel(
    const float* __restrict__ ffn, const float* __restrict__ attn,
    const float* __restrict__ alpha_p, const float* __restrict__ beta_p,
    float* __restrict__ out, int n) {
  int i = blockIdx.x * blockDim.x + threadIdx.x;
  if (i >= n) return;
  float sa = 1.0f / (1.0f + __expf(-alpha_p[0]));
  float sb = 1.0f / (1.0f + __expf(-beta_p[0]));
  out[i] = sa * ffn[i] + sb * attn[i];
}

// ---------------------------------------------------------------------------
extern "C" void kernel_launch(void* const* d_in, const int* in_sizes, int n_in,
                              void* d_out, int out_size, void* d_ws, size_t ws_size,
                              hipStream_t stream) {
  const float* x      = (const float*)d_in[0];
  const float* bias   = (const float*)d_in[1];
  const float* norm_w = (const float*)d_in[2];
  const float* wq     = (const float*)d_in[3];
  const float* wk     = (const float*)d_in[4];
  const float* wv     = (const float*)d_in[5];
  const float* wo     = (const float*)d_in[6];
  const float* w1     = (const float*)d_in[7];
  const float* b1     = (const float*)d_in[8];
  const float* w2     = (const float*)d_in[9];
  const float* b2     = (const float*)d_in[10];
  const float* alpha  = (const float*)d_in[11];
  const float* beta   = (const float*)d_in[12];
  const float* gamma  = (const float*)d_in[13];

  char* ws = (char*)d_ws;
  size_t off = 0;
  auto carve = [&](size_t bytes) -> void* {
    void* p = ws + off;
    off = (off + bytes + 255) & ~(size_t)255;
    return p;
  };
  const size_t XN = (size_t)NSEQ * DIMM;
  bf16_t* xn_b    = (bf16_t*)carve(XN * 2);
  bf16_t* wqT     = (bf16_t*)carve((size_t)DIMM * DIMM * 2);
  bf16_t* wkT     = (bf16_t*)carve((size_t)DIMM * DIMM * 2);
  bf16_t* wvT     = (bf16_t*)carve((size_t)DIMM * DIMM * 2);
  bf16_t* woT     = (bf16_t*)carve((size_t)DIMM * DIMM * 2);
  bf16_t* w1T     = (bf16_t*)carve((size_t)DIMM * DIMM * 2);
  bf16_t* w2T     = (bf16_t*)carve((size_t)DIMM * DIMM * 2);
  bf16_t* Qb      = (bf16_t*)carve(XN * 2);
  bf16_t* Kbf     = (bf16_t*)carve(XN * 2);
  float*  Vf      = (float*) carve(XN * 4);
  bf16_t* attn_b  = (bf16_t*)carve(XN * 2);
  float*  attnout = (float*) carve(XN * 4);
  bf16_t* h1b     = (bf16_t*)carve(XN * 2);
  float*  ffn     = (float*) carve(XN * 4);

  float* out_f   = (float*)d_out;
  int*   idx_out = (int*)d_out + XN;   // second tuple output: topk_idx [1,8,4096,32] i32

  // 1) weights -> transposed bf16
  {
    int n = DIMM * DIMM, blk = 256, grd = (n + blk - 1) / blk;
    transpose_bf16_kernel<<<grd, blk, 0, stream>>>(wq, wqT, DIMM, DIMM);
    transpose_bf16_kernel<<<grd, blk, 0, stream>>>(wk, wkT, DIMM, DIMM);
    transpose_bf16_kernel<<<grd, blk, 0, stream>>>(wv, wvT, DIMM, DIMM);
    transpose_bf16_kernel<<<grd, blk, 0, stream>>>(wo, woT, DIMM, DIMM);
    transpose_bf16_kernel<<<grd, blk, 0, stream>>>(w1, w1T, DIMM, DIMM);
    transpose_bf16_kernel<<<grd, blk, 0, stream>>>(w2, w2T, DIMM, DIMM);
  }

  // 2) sigmoid-scaled RMSNorm
  rmsnorm_kernel<<<NSEQ, DIMM, 0, stream>>>(x, norm_w, gamma, xn_b);

  // 3) Q/K/V projections (WMMA), 32x64 per wave -> 512 waves -> 64 blocks
  int gblks = ((NSEQ / 32) * (DIMM / 64)) / 8;
  gemm_bf16_kernel<<<gblks, 256, 0, stream>>>(xn_b, wqT, nullptr, nullptr, Qb,
                                              NSEQ, DIMM, DIMM, 0);
  gemm_bf16_kernel<<<gblks, 256, 0, stream>>>(xn_b, wkT, nullptr, nullptr, Kbf,
                                              NSEQ, DIMM, DIMM, 0);
  gemm_bf16_kernel<<<gblks, 256, 0, stream>>>(xn_b, wvT, nullptr, Vf, nullptr,
                                              NSEQ, DIMM, DIMM, 0);

  // 4) fused scores + top-k + softmax + gather: 8 heads x 128 row-blocks = 1024 waves
  attn_topk_kernel<<<(NHEAD * (NSEQ / 32)) / 4, 128, 0, stream>>>(
      Qb, Kbf, Vf, bias, attn_b, idx_out);

  // 5) attn @ wo
  gemm_bf16_kernel<<<gblks, 256, 0, stream>>>(attn_b, woT, nullptr, attnout, nullptr,
                                              NSEQ, DIMM, DIMM, 0);

  // 6) FFN: gelu(xn@w1+b1)@w2+b2
  gemm_bf16_kernel<<<gblks, 256, 0, stream>>>(xn_b, w1T, b1, nullptr, h1b,
                                              NSEQ, DIMM, DIMM, 1);
  gemm_bf16_kernel<<<gblks, 256, 0, stream>>>(h1b, w2T, b2, ffn, nullptr,
                                              NSEQ, DIMM, DIMM, 0);

  // 7) blend
  combine_kernel<<<(int)(XN / 256), 256, 0, stream>>>(ffn, attnout, alpha, beta,
                                                      out_f, (int)XN);
}
